// Model_TCN_SNN_74217034875393
// MI455X (gfx1250) — matrix operand order repaired and checked by reading, not compiled, not run on hardware
//
#include <hip/hip_runtime.h>
#include <hip/hip_bf16.h>
#include <math.h>
#include <stdint.h>

// ---------------------------------------------------------------------------
// CDNA5 (gfx1250) implementation of the TCN+SNN+attention model.
// - Dilated convs: tap-decomposed bf16 WMMA implicit-GEMM, f32 accumulate.
//   LDS activation tile stored channel-contiguous so B-fragments are two
//   ds_load_b128 per WMMA; fragment loads double-buffered (2x unrolled) so
//   the loads for iteration i+1 overlap the WMMA of iteration i with no
//   register-rotation moves.
// - SNN scan: TDM (tensor_load_to_lds) stages [16 x 192] f32 tiles into a
//   padded LDS buffer (pad_interval reproduces the [192][17] layout).
// ---------------------------------------------------------------------------

typedef __bf16        v16bf __attribute__((ext_vector_type(16)));
typedef float         v8f   __attribute__((ext_vector_type(8)));
typedef unsigned int  u32x4 __attribute__((ext_vector_type(4)));
typedef int           i32x4 __attribute__((ext_vector_type(4)));
typedef int           i32x8 __attribute__((ext_vector_type(8)));

static constexpr int BB = 64;     // batch
static constexpr int CC = 264;    // input channels
static constexpr int TT = 2000;   // time steps (125 * 16)
static constexpr int HH = 192;    // hidden channels
static constexpr float EPS = 1e-5f;
static constexpr float TANH1 = 0.7615941559557649f;

#if defined(__has_builtin)
#if __has_builtin(__builtin_amdgcn_tensor_load_to_lds)
#define HAS_TDM 1
#endif
#endif
#ifndef HAS_TDM
#define HAS_TDM 0
#endif

// ---------------------------------------------------------------------------
// f32 -> bf16 conversion (activations)
// ---------------------------------------------------------------------------
__global__ void f32_to_bf16_kernel(const float* __restrict__ in,
                                   __bf16* __restrict__ out, long n) {
  long i = (long)blockIdx.x * blockDim.x + threadIdx.x;
  if (i < n) out[i] = (__bf16)in[i];
}

// ---------------------------------------------------------------------------
// Pack conv weights [rows, Cin, KW] (f32) into per-tap WMMA A-fragments:
// dst[mtile][tap][chunk][lane(32)][elem(16)], K = channel, zero-padded to 32.
// A 16x32 bf16 layout as packed here: lanes 0-15 hold M=lane, K in {0..15};
// lanes 16-31 hold M=lane-16, K in {16..31}.
// ---------------------------------------------------------------------------
__global__ void pack_weights_kernel(const float* __restrict__ w,
                                    __bf16* __restrict__ dst,
                                    int mtiles, int cin, int kw, int chunks) {
  int i = blockIdx.x * blockDim.x + threadIdx.x;
  int total = mtiles * kw * chunks * 512;
  if (i >= total) return;
  int e     = i & 15;
  int lane  = (i >> 4) & 31;
  int rest  = i >> 9;
  int chunk = rest % chunks;  rest /= chunks;
  int tap   = rest % kw;
  int mtile = rest / kw;
  int k   = chunk * 32 + e + ((lane >> 4) << 4);
  int row = mtile * 16 + (lane & 15);
  float val = 0.f;
  if (k < cin) val = w[(row * cin + k) * kw + tap];
  dst[i] = (__bf16)val;
}

// ---------------------------------------------------------------------------
// Per-wave WMMA accumulation: sum over taps of 32-channel-chunk GEMMs,
// flattened to one iteration space. Double-buffered 2x unroll: fragments for
// iteration i+1 load into the idle buffer while the WMMA of iteration i
// consumes the other -- no rotation moves, partial counter waits at the WMMA.
// ---------------------------------------------------------------------------
template <int KW, int CIN, int CINP, int DIL>
__device__ __forceinline__ v8f accum_conv(v8f c,
                                          const __bf16* __restrict__ pack,
                                          const __bf16* tile, int lane) {
  constexpr int CH   = (CIN + 31) / 32;
  constexpr int JOFF = 3 - (KW - 1) / 2;   // tap-center alignment in halo tile
  constexpr int NIT  = KW * CH;
  const int n = lane & 15;
  const int khalf = lane >> 4;
  const __bf16* ap   = pack + lane * 16;
  const __bf16* brow = tile + n * CINP + khalf * 16 + JOFF * DIL * CINP;
  int ch = 0;
  auto advance = [&]() {
    ap += 512;
    __builtin_prefetch(ap + 512, 0, 1);
    if (++ch == CH) { ch = 0; brow += DIL * CINP; }   // next tap row
  };
  v16bf a0 = *(const v16bf*)ap;            // prologue: fragments for i=0
  v16bf b0 = *(const v16bf*)brow;
#pragma unroll 1
  for (int i = 0; i < NIT; i += 2) {       // guards are wave-uniform
    v16bf a1, b1;
    if (i + 1 < NIT) {
      advance();
      a1 = *(const v16bf*)ap;
      b1 = *(const v16bf*)(brow + ch * 32);
    }
    c = __builtin_amdgcn_wmma_f32_16x16x32_bf16(
        false, a0, false, b0, (short)0, c, false, false);
    if (i + 1 < NIT) {
      if (i + 2 < NIT) {
        advance();
        a0 = *(const v16bf*)ap;
        b0 = *(const v16bf*)(brow + ch * 32);
      }
      c = __builtin_amdgcn_wmma_f32_16x16x32_bf16(
          false, a1, false, b1, (short)0, c, false, false);
    }
  }
  return c;
}

// ---------------------------------------------------------------------------
// One TCN block: 12 waves per workgroup, wave w owns out-ch tile w.
// Tiles 0-3: k=3 conv; 4-7: k=5; 8-11: k=7 (matches channel concat order).
// Fused: bias + BN + ReLU + residual (block1: 1x1 conv GEMM; else: identity).
// LDS tile layout: tile[j][ci] (channel-contiguous), padded channels zeroed.
// ---------------------------------------------------------------------------
template <int CIN, int DIL, bool FIRST>
__global__ __launch_bounds__(384)
void tcn_block_kernel(const __bf16* __restrict__ xb,       // [B,CIN,T] bf16
                      const float*  __restrict__ resid,    // [B,H,T] f32 (if !FIRST)
                      const __bf16* __restrict__ p3, const __bf16* __restrict__ p5,
                      const __bf16* __restrict__ p7, const __bf16* __restrict__ prw,
                      const float* __restrict__ bias3, const float* __restrict__ bias5,
                      const float* __restrict__ bias7, const float* __restrict__ rbias,
                      const float* __restrict__ g,  const float* __restrict__ bbeta,
                      const float* __restrict__ mu, const float* __restrict__ var,
                      float* __restrict__ outf, __bf16* __restrict__ outb) {
  constexpr int TILEW = 16 + 6 * DIL;     // 16 cols + k=7 halo both sides
  constexpr int CH    = (CIN + 31) / 32;  // 32-channel K chunks
  constexpr int CSTG  = CH * 32;          // staged channels (zero padded)
  constexpr int CINP  = CSTG + 8;         // +8 bf16 row pad for LDS banks

  __shared__ __bf16 tile[TILEW * CINP];

  const int b   = blockIdx.y;
  const int t0  = blockIdx.x * 16;
  const int tid = threadIdx.x;
  const int ts  = t0 - 3 * DIL;

  // Stage haloed tile, transposing to [time][channel]; global reads stay
  // t-coalesced, scatter happens on the LDS-store side.
  for (int i = tid; i < TILEW * CSTG; i += 384) {
    int ci = i / TILEW, j = i - ci * TILEW;
    int t = ts + j;
    __bf16 v = (__bf16)0.f;
    if (ci < CIN && t >= 0 && t < TT) v = xb[((size_t)b * CIN + ci) * TT + t];
    tile[j * CINP + ci] = v;
  }
  __syncthreads();

  const int wave = tid >> 5, lane = tid & 31;
  const int which = wave >> 2, mloc = wave & 3;

  v8f c = {0.f, 0.f, 0.f, 0.f, 0.f, 0.f, 0.f, 0.f};
  if (which == 0)
    c = accum_conv<3, CIN, CINP, DIL>(c, p3 + (size_t)mloc * (3 * CH) * 512, tile, lane);
  else if (which == 1)
    c = accum_conv<5, CIN, CINP, DIL>(c, p5 + (size_t)mloc * (5 * CH) * 512, tile, lane);
  else
    c = accum_conv<7, CIN, CINP, DIL>(c, p7 + (size_t)mloc * (7 * CH) * 512, tile, lane);

  v8f cr = {0.f, 0.f, 0.f, 0.f, 0.f, 0.f, 0.f, 0.f};
  if (FIRST)   // 1x1 residual projection (192 x CIN GEMM), one M-tile per wave
    cr = accum_conv<1, CIN, CINP, DIL>(cr, prw + (size_t)wave * CH * 512, tile, lane);

  // Epilogue. C/D layout: VGPR r -> row (r + 8*(lane>=16)), col = lane&15.
  const int n = lane & 15, khalf = lane >> 4;
  const int t = t0 + n;
  const float* biasArr = (which == 0) ? bias3 : (which == 1) ? bias5 : bias7;
#pragma unroll
  for (int r = 0; r < 8; ++r) {
    int ocl = mloc * 16 + khalf * 8 + r;   // row within this conv (0..63)
    int oc  = wave * 16 + khalf * 8 + r;   // concatenated channel (0..191)
    float y  = c[r] + biasArr[ocl];
    float yb = (y - mu[oc]) * g[oc] * rsqrtf(var[oc] + EPS) + bbeta[oc];
    yb = fmaxf(yb, 0.f);
    float res;
    if (FIRST) res = cr[r] + rbias[oc];
    else       res = resid[((size_t)b * HH + oc) * TT + t];
    float o = yb + res;
    size_t oi = ((size_t)b * HH + oc) * TT + t;
    outf[oi] = o;
    outb[oi] = (__bf16)o;
  }
}

// ---------------------------------------------------------------------------
// SNN: BN + PLIF leaky-integrate-fire scan over T (sequential), emitting
// spikes [B,T,H] and per-timestep attention scores (tanh(1)*dot + bias).
// One block per batch, thread = hidden channel; 16-step tiles staged by TDM.
// ---------------------------------------------------------------------------
__global__ __launch_bounds__(192)
void snn_scan_kernel(const float* __restrict__ h3, const float* __restrict__ plif,
                     const float* __restrict__ g,  const float* __restrict__ bbeta,
                     const float* __restrict__ mu, const float* __restrict__ var,
                     const float* __restrict__ attn_w, const float* __restrict__ attn_b,
                     float* __restrict__ spk, float* __restrict__ scores) {
  const int b = blockIdx.x, h = threadIdx.x;
  __shared__ float xt[HH][17];   // 17 = 16 data + 1 pad (== TDM pad layout)
  __shared__ float red[16];

  const float decay = 1.f / (1.f + __expf(-plif[0]));
  const float sc = g[h] * rsqrtf(var[h] + EPS);
  const float mh = mu[h], bh = bbeta[h], aw = attn_w[h];
  const float biasv = attn_b[0];
  float v = 0.f;

  const uint64_t gbase = (uint64_t)(uintptr_t)(h3 + (size_t)b * HH * TT);
  const uint32_t lds0  = (uint32_t)(uintptr_t)(&xt[0][0]);
  (void)gbase; (void)lds0;

  for (int t0 = 0; t0 < TT; t0 += 16) {
    __syncthreads();               // previous tile fully consumed
#if HAS_TDM
    if (threadIdx.x < 32) {        // wave 0 issues the TDM descriptor
      uint64_t ga = gbase + (uint64_t)t0 * 4u;
      u32x4 g0;
      g0.x = 1u;                                   // count=1, user mode
      g0.y = lds0;                                 // lds_addr
      g0.z = (uint32_t)ga;                         // global_addr[31:0]
      g0.w = ((uint32_t)(ga >> 32) & 0x01FFFFFFu)  // global_addr[56:32]
             | (2u << 30);                         // type = image
      i32x8 g1;
      g1[0] = (int)((2u << 16)      // data_size = 4B
                  | (1u << 20)      // pad_enable
                  | (3u << 22)      // pad_interval: every 16 DWORDs
                  | (0u << 25));    // pad_amount: 1 DWORD  -> row stride 17
      g1[1] = (int)((uint32_t)TT << 16);   // tensor_dim0 = 2000 (lo16)
      g1[2] = (int)((uint32_t)HH << 16);   // dim0 hi=0, tensor_dim1 = 192
      g1[3] = (int)(16u << 16);            // dim1 hi=0, tile_dim0 = 16
      g1[4] = (int)(uint32_t)HH;           // tile_dim1 = 192, tile_dim2 = 0
      g1[5] = (int)(uint32_t)TT;           // tensor_dim0_stride = 2000 (lo32)
      g1[6] = 0;                           // stride0 hi, stride1 lo
      g1[7] = 0;
      i32x4 gz4 = {0, 0, 0, 0};
      i32x8 gz8 = {0, 0, 0, 0, 0, 0, 0, 0};
      // 6-arg form (clang-23 / therock-10.0): (g0, g1, g2, g3, g4, cpol)
      __builtin_amdgcn_tensor_load_to_lds(g0, g1, gz4, gz4, gz8, 0);
      __builtin_amdgcn_s_wait_tensorcnt(0);
    }
#else
    for (int i = 0; i < 16; ++i) {  // fallback: coalesced manual staging
      int flat = h + i * HH;
      int hh = flat >> 4, j = flat & 15;
      xt[hh][j] = h3[((size_t)b * HH + hh) * TT + t0 + j];
    }
#endif
    if (h < 16) red[h] = 0.f;
    __syncthreads();

    for (int j = 0; j < 16; ++j) {
      float x = (xt[h][j] - mh) * sc + bh;
      v += (x - v) * decay;               // decay_input, v_reset=0
      float s = (v >= 1.f) ? 1.f : 0.f;   // v_threshold = 1
      v *= (1.f - s);                     // hard reset
      spk[((size_t)b * TT + t0 + j) * HH + h] = s;
      if (s != 0.f) atomicAdd(&red[j], aw);
    }
    __syncthreads();
    if (h < 16) scores[b * TT + t0 + h] = TANH1 * red[h] + biasv;
  }
}

// ---------------------------------------------------------------------------
// Attention softmax over T + weighted pooling -> pooled [B,H].
// ---------------------------------------------------------------------------
__global__ __launch_bounds__(256)
void attn_pool_kernel(const float* __restrict__ scores,
                      const float* __restrict__ spk,
                      float* __restrict__ pooled) {
  const int b = blockIdx.x, tid = threadIdx.x;
  __shared__ float wbuf[TT];
  __shared__ float red[256];

  float lm = -1e30f;
  for (int t = tid; t < TT; t += 256) lm = fmaxf(lm, scores[b * TT + t]);
  red[tid] = lm; __syncthreads();
  for (int s = 128; s > 0; s >>= 1) {
    if (tid < s) red[tid] = fmaxf(red[tid], red[tid + s]);
    __syncthreads();
  }
  float mx = red[0]; __syncthreads();

  float ls = 0.f;
  for (int t = tid; t < TT; t += 256) {
    float e = __expf(scores[b * TT + t] - mx);
    wbuf[t] = e; ls += e;
  }
  red[tid] = ls; __syncthreads();
  for (int s = 128; s > 0; s >>= 1) {
    if (tid < s) red[tid] += red[tid + s];
    __syncthreads();
  }
  float inv = 1.f / red[0];
  __syncthreads();

  if (tid < HH) {
    float acc = 0.f;
    for (int t = 0; t < TT; ++t)
      acc += spk[((size_t)b * TT + t) * HH + tid] * wbuf[t];
    pooled[b * HH + tid] = acc * inv;
  }
}

// ---------------------------------------------------------------------------
// MLP head: 192 -> 128 (BN+GELU) -> 64 (BN+GELU) -> 4. One block per batch.
// ---------------------------------------------------------------------------
__global__ __launch_bounds__(192)
void head_kernel(const float* __restrict__ pooled,
                 const float* c1w, const float* c1b, const float* c1g,
                 const float* c1bb, const float* c1m, const float* c1v,
                 const float* c2w, const float* c2b, const float* c2g,
                 const float* c2bb, const float* c2m, const float* c2v,
                 const float* c3w, const float* c3b,
                 float* __restrict__ out) {
  const int b = blockIdx.x, tid = threadIdx.x;
  __shared__ float sp[HH], s1[128], s2[64];
  if (tid < HH) sp[tid] = pooled[b * HH + tid];
  __syncthreads();
  if (tid < 128) {
    float a = c1b[tid];
    for (int i = 0; i < HH; ++i) a += sp[i] * c1w[tid * HH + i];
    a = (a - c1m[tid]) * c1g[tid] * rsqrtf(c1v[tid] + EPS) + c1bb[tid];
    s1[tid] = 0.5f * a * (1.f + erff(a * 0.70710678118654752f));
  }
  __syncthreads();
  if (tid < 64) {
    float a = c2b[tid];
    for (int i = 0; i < 128; ++i) a += s1[i] * c2w[tid * 128 + i];
    a = (a - c2m[tid]) * c2g[tid] * rsqrtf(c2v[tid] + EPS) + c2bb[tid];
    s2[tid] = 0.5f * a * (1.f + erff(a * 0.70710678118654752f));
  }
  __syncthreads();
  if (tid < 4) {
    float a = c3b[tid];
    for (int i = 0; i < 64; ++i) a += s2[i] * c3w[tid * 64 + i];
    out[b * 4 + tid] = a;
  }
}

// ---------------------------------------------------------------------------
// Host launch
// ---------------------------------------------------------------------------
extern "C" void kernel_launch(void* const* d_in, const int* in_sizes, int n_in,
                              void* d_out, int out_size, void* d_ws, size_t ws_size,
                              hipStream_t stream) {
  // ---- inputs (setup_inputs dict order) ----
  const float* x     = (const float*)d_in[0];
  const float* t1w[3] = {(const float*)d_in[1], (const float*)d_in[3], (const float*)d_in[5]};
  const float* t1b[3] = {(const float*)d_in[2], (const float*)d_in[4], (const float*)d_in[6]};
  const float *t1g = (const float*)d_in[7],  *t1bb = (const float*)d_in[8],
              *t1m = (const float*)d_in[9],  *t1v  = (const float*)d_in[10];
  const float* t2w[3] = {(const float*)d_in[11], (const float*)d_in[13], (const float*)d_in[15]};
  const float* t2b[3] = {(const float*)d_in[12], (const float*)d_in[14], (const float*)d_in[16]};
  const float *t2g = (const float*)d_in[17], *t2bb = (const float*)d_in[18],
              *t2m = (const float*)d_in[19], *t2v  = (const float*)d_in[20];
  const float* t3w[3] = {(const float*)d_in[21], (const float*)d_in[23], (const float*)d_in[25]};
  const float* t3b[3] = {(const float*)d_in[22], (const float*)d_in[24], (const float*)d_in[26]};
  const float *t3g = (const float*)d_in[27], *t3bb = (const float*)d_in[28],
              *t3m = (const float*)d_in[29], *t3v  = (const float*)d_in[30];
  const float *t1rw = (const float*)d_in[31], *t1rb = (const float*)d_in[32];
  const float *sg = (const float*)d_in[33], *sbb = (const float*)d_in[34],
              *sm = (const float*)d_in[35], *sv  = (const float*)d_in[36];
  const float *plif = (const float*)d_in[37];
  const float *aw = (const float*)d_in[38], *ab = (const float*)d_in[39];
  const float *c1w = (const float*)d_in[40], *c1b = (const float*)d_in[41],
              *c1g = (const float*)d_in[42], *c1bb = (const float*)d_in[43],
              *c1m = (const float*)d_in[44], *c1v = (const float*)d_in[45];
  const float *c2w = (const float*)d_in[46], *c2b = (const float*)d_in[47],
              *c2g = (const float*)d_in[48], *c2bb = (const float*)d_in[49],
              *c2m = (const float*)d_in[50], *c2v = (const float*)d_in[51];
  const float *c3w = (const float*)d_in[52], *c3b = (const float*)d_in[53];

  // ---- workspace layout (explicit aliasing to bound footprint ~415 MB) ----
  char* ws = (char*)d_ws;
  size_t off = 0;
  auto take = [&](size_t bytes) { size_t r = off; off += (bytes + 255) & ~(size_t)255; return r; };
  const size_t oXB  = take((size_t)BB * CC * TT * 2);   // x bf16; later reused as h2 bf16
  const size_t oH1  = take((size_t)BB * HH * TT * 4);   // h1 f32; later reused as spikes
  const size_t oH1B = take((size_t)BB * HH * TT * 2);   // h1 bf16; later reused as h3-bf16 dump
  const size_t oH2  = take((size_t)BB * HH * TT * 4);
  const size_t oH3  = take((size_t)BB * HH * TT * 4);
  const size_t oSC  = take((size_t)BB * TT * 4);
  const size_t oPL  = take((size_t)BB * HH * 4);
  auto packBytes = [](int mtiles, int cin, int kw) {
    return (size_t)mtiles * kw * ((cin + 31) / 32) * 512 * 2;
  };
  const size_t oP13 = take(packBytes(4, CC, 3));
  const size_t oP15 = take(packBytes(4, CC, 5));
  const size_t oP17 = take(packBytes(4, CC, 7));
  const size_t oP1R = take(packBytes(12, CC, 1));
  const size_t oP23 = take(packBytes(4, HH, 3));
  const size_t oP25 = take(packBytes(4, HH, 5));
  const size_t oP27 = take(packBytes(4, HH, 7));
  const size_t oP33 = take(packBytes(4, HH, 3));
  const size_t oP35 = take(packBytes(4, HH, 5));
  const size_t oP37 = take(packBytes(4, HH, 7));
  (void)ws_size; (void)in_sizes; (void)n_in; (void)out_size;

  __bf16* xb  = (__bf16*)(ws + oXB);
  float*  h1  = (float*)(ws + oH1);
  __bf16* h1b = (__bf16*)(ws + oH1B);
  float*  h2  = (float*)(ws + oH2);
  __bf16* h2b = (__bf16*)(ws + oXB);      // alias: xb dead after block1
  float*  h3  = (float*)(ws + oH3);
  __bf16* h3bDump = (__bf16*)(ws + oH1B); // alias: h1b dead after block2
  float*  spk = (float*)(ws + oH1);       // alias: h1 dead after block2
  float*  sc  = (float*)(ws + oSC);
  float*  pl  = (float*)(ws + oPL);

  // 1) activations -> bf16
  {
    long n = (long)BB * CC * TT;
    f32_to_bf16_kernel<<<(n + 255) / 256, 256, 0, stream>>>(x, xb, n);
  }
  // 2) weight packs (per-tap A-fragment layout)
  auto pack = [&](const float* w, size_t o, int mt, int cin, int kw) {
    int ch = (cin + 31) / 32;
    int total = mt * kw * ch * 512;
    pack_weights_kernel<<<(total + 255) / 256, 256, 0, stream>>>(
        w, (__bf16*)(ws + o), mt, cin, kw, ch);
  };
  pack(t1w[0], oP13, 4, CC, 3);  pack(t1w[1], oP15, 4, CC, 5);  pack(t1w[2], oP17, 4, CC, 7);
  pack(t1rw,  oP1R, 12, CC, 1);
  pack(t2w[0], oP23, 4, HH, 3);  pack(t2w[1], oP25, 4, HH, 5);  pack(t2w[2], oP27, 4, HH, 7);
  pack(t3w[0], oP33, 4, HH, 3);  pack(t3w[1], oP35, 4, HH, 5);  pack(t3w[2], oP37, 4, HH, 7);

  dim3 grid(TT / 16, BB);
  // 3) TCN block 1 (dil=1, conv residual)
  tcn_block_kernel<CC, 1, true><<<grid, 384, 0, stream>>>(
      xb, nullptr,
      (const __bf16*)(ws + oP13), (const __bf16*)(ws + oP15),
      (const __bf16*)(ws + oP17), (const __bf16*)(ws + oP1R),
      t1b[0], t1b[1], t1b[2], t1rb, t1g, t1bb, t1m, t1v, h1, h1b);
  // 4) TCN block 2 (dil=2, identity residual)
  tcn_block_kernel<HH, 2, false><<<grid, 384, 0, stream>>>(
      h1b, h1,
      (const __bf16*)(ws + oP23), (const __bf16*)(ws + oP25),
      (const __bf16*)(ws + oP27), nullptr,
      t2b[0], t2b[1], t2b[2], nullptr, t2g, t2bb, t2m, t2v, h2, h2b);
  // 5) TCN block 3 (dil=4, identity residual)
  tcn_block_kernel<HH, 4, false><<<grid, 384, 0, stream>>>(
      h2b, h2,
      (const __bf16*)(ws + oP33), (const __bf16*)(ws + oP35),
      (const __bf16*)(ws + oP37), nullptr,
      t3b[0], t3b[1], t3b[2], nullptr, t3g, t3bb, t3m, t3v, h3, h3bDump);
  // 6) SNN scan + attention scores (TDM-staged tiles)
  snn_scan_kernel<<<BB, HH, 0, stream>>>(h3, plif, sg, sbb, sm, sv, aw, ab, spk, sc);
  // 7) softmax pooling
  attn_pool_kernel<<<BB, 256, 0, stream>>>(sc, spk, pl);
  // 8) MLP head -> d_out [64,4] f32
  head_kernel<<<BB, HH, 0, stream>>>(pl, c1w, c1b, c1g, c1bb, c1m, c1v,
                                     c2w, c2b, c2g, c2bb, c2m, c2v,
                                     c3w, c3b, (float*)d_out);
}
